// TorchLMHeadCE_69217692942469
// MI455X (gfx1250) — matrix operand order repaired and checked by reading, not compiled, not run on hardware
//
#include <hip/hip_runtime.h>

#define H_DIM 4096
#define IGNORE_INDEX (-100)
#define NEG_BIG (-3.0e38f)

#define BM 32          // rows per block (2 M-tiles)
#define BN 256         // cols per block chunk (8 waves * 2 N-tiles * 16)
#define KSS 512        // K super-step staged in LDS
#define VSPLIT 5       // independent V partitions (32000/5 = 6400 = 25*256)

typedef __bf16 bf16_t;
typedef bf16_t v16bf __attribute__((ext_vector_type(16)));
typedef float  v8f   __attribute__((ext_vector_type(8)));
typedef unsigned int v8u __attribute__((ext_vector_type(8)));

// Pack two fp32 into one dword of bf16 (truncation) with a single v_perm_b32.
__device__ __forceinline__ unsigned pack_bf2(float lo, float hi) {
  return __builtin_amdgcn_perm(__builtin_bit_cast(unsigned, hi),
                               __builtin_bit_cast(unsigned, lo),
                               0x07060302u);
}

// B fragment (32x16 bf16) from global fp32: 16 consecutive K values per lane.
__device__ __forceinline__ v16bf load_b_frag_f32(const float* __restrict__ p) {
  const float4 g0 = *(const float4*)(p);
  const float4 g1 = *(const float4*)(p + 4);
  const float4 g2 = *(const float4*)(p + 8);
  const float4 g3 = *(const float4*)(p + 12);
  v8u bu;
  bu[0] = pack_bf2(g0.x, g0.y); bu[1] = pack_bf2(g0.z, g0.w);
  bu[2] = pack_bf2(g1.x, g1.y); bu[3] = pack_bf2(g1.z, g1.w);
  bu[4] = pack_bf2(g2.x, g2.y); bu[5] = pack_bf2(g2.z, g2.w);
  bu[6] = pack_bf2(g3.x, g3.y); bu[7] = pack_bf2(g3.z, g3.w);
  return __builtin_bit_cast(v16bf, bu);
}

// B fragment from pre-converted bf16 W: two global_load_b128, no VALU.
__device__ __forceinline__ v16bf load_b_frag_h(const unsigned short* __restrict__ p) {
  const uint4 lo = *(const uint4*)(p);       // K = bkoff .. bkoff+7
  const uint4 hi = *(const uint4*)(p + 8);   // K = bkoff+8 .. bkoff+15
  v8u bu;
  bu[0] = lo.x; bu[1] = lo.y; bu[2] = lo.z; bu[3] = lo.w;
  bu[4] = hi.x; bu[5] = hi.y; bu[6] = hi.z; bu[7] = hi.w;
  return __builtin_bit_cast(v16bf, bu);
}

// A fragment (16x32 bf16) from pre-converted LDS: two ds_load_b128, no VALU.
__device__ __forceinline__ v16bf load_a_frag(const unsigned short* ap) {
  const uint4 lo = *(const uint4*)(ap);        // K = koff .. koff+7
  const uint4 hi = *(const uint4*)(ap + 8);    // K = koff+8 .. koff+15
  v8u au;
  au[0] = lo.x; au[1] = lo.y; au[2] = lo.z; au[3] = lo.w;
  au[4] = hi.x; au[5] = hi.y; au[6] = hi.z; au[7] = hi.w;
  return __builtin_bit_cast(v16bf, au);
}

// Streaming fp32 -> bf16 conversion of W (one pass, fully coalesced).
__global__ __launch_bounds__(256)
void convert_w_bf16(const float* __restrict__ src, unsigned short* __restrict__ dst,
                    long long total) {
  const long long stride = (long long)gridDim.x * blockDim.x;
  for (long long i = (long long)blockIdx.x * blockDim.x + threadIdx.x;
       i * 16 < total; i += stride) {
    const long long e = i * 16;
    const float4 f0 = *(const float4*)(src + e);
    const float4 f1 = *(const float4*)(src + e + 4);
    const float4 f2 = *(const float4*)(src + e + 8);
    const float4 f3 = *(const float4*)(src + e + 12);
    uint4 p0, p1;
    p0.x = pack_bf2(f0.x, f0.y); p0.y = pack_bf2(f0.z, f0.w);
    p0.z = pack_bf2(f1.x, f1.y); p0.w = pack_bf2(f1.z, f1.w);
    p1.x = pack_bf2(f2.x, f2.y); p1.y = pack_bf2(f2.z, f2.w);
    p1.z = pack_bf2(f3.x, f3.y); p1.w = pack_bf2(f3.z, f3.w);
    *(uint4*)(dst + e)     = p0;
    *(uint4*)(dst + e + 8) = p1;
  }
}

template <bool WBF16>
__global__ __launch_bounds__(256)
void lmhead_ce_partial(const float* __restrict__ x,
                       const float* __restrict__ Wf,
                       const unsigned short* __restrict__ Wh,
                       const int*   __restrict__ y,
                       float* __restrict__ pm,   // [VSPLIT][N] running max
                       float* __restrict__ ps,   // [VSPLIT][N] running sumexp
                       float* __restrict__ pt,   // [VSPLIT][N] target logit
                       int N, int V, int Hd) {
  constexpr int LDA = KSS + 8;                  // padded row stride (halfwords)
  __shared__ __align__(16) unsigned short xs[BM * LDA];
  __shared__ int   ytile[BM];
  __shared__ float red_m[8][BM];
  __shared__ float red_s[8][BM];
  __shared__ float red_t[8][BM];

  const int tid  = threadIdx.x;
  const int lane = tid & 31;
  const int wave = tid >> 5;
  const int r0   = blockIdx.x * BM;
  const int vlen = V / VSPLIT;                  // 6400
  const int c0   = blockIdx.y * vlen;

  if (tid < BM) ytile[tid] = y[r0 + tid];

  const int arow   = lane & 15;
  const int akoff  = (lane >> 4) << 4;          // 0 or 16
  const int bcol   = lane & 15;
  const int bkoff  = (lane >> 4) << 4;          // 0 or 16
  const int mybase = (lane >> 4) << 3;          // acc rows 0..7 or 8..15

  // cooperative A staging: thread -> (row, 64-col segment)
  const int srow = tid >> 3;
  const int sseg = (tid & 7) << 6;
  const float* sx = x + (size_t)(r0 + srow) * Hd + sseg;
  unsigned short* sdst = &xs[srow * LDA + sseg];

  float runm[16], runs[16], tgt[16];
  #pragma unroll
  for (int i = 0; i < 16; ++i) { runm[i] = NEG_BIG; runs[i] = 0.0f; tgt[i] = NEG_BIG; }

  const int nchunks = vlen / BN;                // 25

  for (int vc = 0; vc < nchunks; ++vc) {
    const int v0 = c0 + vc * BN + wave * 32;    // this wave's first column
    const float*          wf0 = Wf + (size_t)(v0 + bcol) * Hd + bkoff;
    const float*          wf1 = wf0 + (size_t)16 * Hd;
    const unsigned short* wh0 = Wh + (size_t)(v0 + bcol) * Hd + bkoff;
    const unsigned short* wh1 = wh0 + (size_t)16 * Hd;

    v8f c00 = {}, c01 = {}, c10 = {}, c11 = {};

    for (int ks0 = 0; ks0 < Hd; ks0 += KSS) {
      __syncthreads();                          // protect xs from prior readers
      #pragma unroll
      for (int i = 0; i < 8; ++i) {             // stage 64 floats -> 64 bf16
        const float4 f0 = *(const float4*)(sx + ks0 + i * 8);
        const float4 f1 = *(const float4*)(sx + ks0 + i * 8 + 4);
        uint4 pkd;
        pkd.x = pack_bf2(f0.x, f0.y); pkd.y = pack_bf2(f0.z, f0.w);
        pkd.z = pack_bf2(f1.x, f1.y); pkd.w = pack_bf2(f1.z, f1.w);
        *(uint4*)(sdst + i * 8) = pkd;
      }
      __syncthreads();

      #pragma unroll 2
      for (int kk = 0; kk < KSS; kk += 32) {
        const v16bf a0 = load_a_frag(&xs[(arow)      * LDA + kk + akoff]);
        const v16bf a1 = load_a_frag(&xs[(16 + arow) * LDA + kk + akoff]);
        v16bf b0, b1;
        if constexpr (WBF16) {
          b0 = load_b_frag_h(wh0 + ks0 + kk);
          b1 = load_b_frag_h(wh1 + ks0 + kk);
        } else {
          b0 = load_b_frag_f32(wf0 + ks0 + kk);
          b1 = load_b_frag_f32(wf1 + ks0 + kk);
        }
        c00 = __builtin_amdgcn_wmma_f32_16x16x32_bf16(false, a0, false, b0, (short)0, c00, false, false);
        c01 = __builtin_amdgcn_wmma_f32_16x16x32_bf16(false, a0, false, b1, (short)0, c01, false, false);
        c10 = __builtin_amdgcn_wmma_f32_16x16x32_bf16(false, a1, false, b0, (short)0, c10, false, false);
        c11 = __builtin_amdgcn_wmma_f32_16x16x32_bf16(false, a1, false, b1, (short)0, c11, false, false);
      }
    }

    // online softmax update: each lane sees 2 columns for 16 rows
    const int ncol0 = v0 + bcol;
    const int ncol1 = ncol0 + 16;
    #pragma unroll
    for (int j = 0; j < 8; ++j) {
      {
        const int yrow = ytile[mybase + j];
        float v = c00[j];
        float nm = fmaxf(runm[j], v);
        runs[j] = runs[j] * __expf(runm[j] - nm) + __expf(v - nm);
        runm[j] = nm;
        if (yrow == ncol0) tgt[j] = v;
        v = c01[j];
        nm = fmaxf(runm[j], v);
        runs[j] = runs[j] * __expf(runm[j] - nm) + __expf(v - nm);
        runm[j] = nm;
        if (yrow == ncol1) tgt[j] = v;
      }
      {
        const int si = 8 + j;
        const int yrow = ytile[16 + mybase + j];
        float v = c10[j];
        float nm = fmaxf(runm[si], v);
        runs[si] = runs[si] * __expf(runm[si] - nm) + __expf(v - nm);
        runm[si] = nm;
        if (yrow == ncol0) tgt[si] = v;
        v = c11[j];
        nm = fmaxf(runm[si], v);
        runs[si] = runs[si] * __expf(runm[si] - nm) + __expf(v - nm);
        runm[si] = nm;
        if (yrow == ncol1) tgt[si] = v;
      }
    }
  }

  // reduce (max,sum,target) across the 16 lanes (columns) of each half-wave
  #pragma unroll
  for (int d = 1; d < 16; d <<= 1) {
    #pragma unroll
    for (int si = 0; si < 16; ++si) {
      const float om = __shfl_xor(runm[si], d, 16);
      const float os = __shfl_xor(runs[si], d, 16);
      const float ot = __shfl_xor(tgt[si],  d, 16);
      const float nm = fmaxf(runm[si], om);
      runs[si] = runs[si] * __expf(runm[si] - nm) + os * __expf(om - nm);
      runm[si] = nm;
      tgt[si]  = fmaxf(tgt[si], ot);
    }
  }
  if ((lane & 15) == 0) {
    #pragma unroll
    for (int mt = 0; mt < 2; ++mt) {
      #pragma unroll
      for (int j = 0; j < 8; ++j) {
        const int row = mt * 16 + mybase + j;
        red_m[wave][row] = runm[mt * 8 + j];
        red_s[wave][row] = runs[mt * 8 + j];
        red_t[wave][row] = tgt[mt * 8 + j];
      }
    }
  }
  __syncthreads();

  // combine the 8 waves' partials; emit per-(split,row) partial stats
  if (tid < BM) {
    float m = NEG_BIG, s = 0.0f, tg = NEG_BIG;
    #pragma unroll
    for (int w = 0; w < 8; ++w) {
      const float om = red_m[w][tid];
      const float os = red_s[w][tid];
      const float nm = fmaxf(m, om);
      s = s * __expf(m - nm) + os * __expf(om - nm);
      m = nm;
      tg = fmaxf(tg, red_t[w][tid]);
    }
    const int o = blockIdx.y * N + r0 + tid;
    pm[o] = m; ps[o] = s; pt[o] = tg;
  }
}

__global__ __launch_bounds__(256)
void ce_finalize(const float* __restrict__ pm, const float* __restrict__ ps,
                 const float* __restrict__ pt, const int* __restrict__ y,
                 float* __restrict__ out, int N) {
  __shared__ float ssum[256];
  __shared__ int   scnt[256];
  float s = 0.0f; int c = 0;
  for (int i = threadIdx.x; i < N; i += 256) {
    float m = NEG_BIG, su = 0.0f, tg = NEG_BIG;
    #pragma unroll
    for (int v = 0; v < VSPLIT; ++v) {
      const float om = pm[v * N + i];
      const float os = ps[v * N + i];
      const float nm = fmaxf(m, om);
      su = su * __expf(m - nm) + os * __expf(om - nm);
      m = nm;
      tg = fmaxf(tg, pt[v * N + i]);
    }
    if (y[i] != IGNORE_INDEX) { s += (__logf(su) + m) - tg; c += 1; }
  }
  ssum[threadIdx.x] = s; scnt[threadIdx.x] = c;
  __syncthreads();
  for (int d = 128; d > 0; d >>= 1) {
    if (threadIdx.x < d) {
      ssum[threadIdx.x] += ssum[threadIdx.x + d];
      scnt[threadIdx.x] += scnt[threadIdx.x + d];
    }
    __syncthreads();
  }
  if (threadIdx.x == 0) {
    const int cnt = scnt[0] > 1 ? scnt[0] : 1;
    out[0] = ssum[0] / (float)cnt;
  }
}

extern "C" void kernel_launch(void* const* d_in, const int* in_sizes, int n_in,
                              void* d_out, int out_size, void* d_ws, size_t ws_size,
                              hipStream_t stream) {
  const float* x = (const float*)d_in[0];
  const float* W = (const float*)d_in[1];
  const int*   y = (const int*)d_in[2];

  const int N  = in_sizes[2];
  const int Hd = H_DIM;
  const int V  = in_sizes[1] / Hd;
  const long long welts = (long long)V * Hd;

  float* pm = (float*)d_ws;                       // VSPLIT*N floats
  float* ps = pm + (size_t)VSPLIT * N;
  float* pt = ps + (size_t)VSPLIT * N;

  const size_t stats_bytes = (size_t)3 * VSPLIT * N * sizeof(float);
  const size_t woff        = (stats_bytes + 255) & ~(size_t)255;
  const size_t need        = woff + (size_t)welts * sizeof(unsigned short);

  const dim3 grid(N / BM, VSPLIT), block(256);

  if (ws_size >= need) {
    // Fast path: one streaming pass converts W to bf16, then the GEMM inner
    // loop is pure b128 loads + WMMA (no per-use conversion VALU).
    unsigned short* Wh = (unsigned short*)((char*)d_ws + woff);
    convert_w_bf16<<<dim3(16384), block, 0, stream>>>(W, Wh, welts);
    lmhead_ce_partial<true><<<grid, block, 0, stream>>>(x, W, Wh, y, pm, ps, pt, N, V, Hd);
  } else {
    lmhead_ce_partial<false><<<grid, block, 0, stream>>>(x, W, nullptr, y, pm, ps, pt, N, V, Hd);
  }
  ce_finalize<<<dim3(1), block, 0, stream>>>(pm, ps, pt, y, (float*)d_out, N);
}
